// AsyncNaiveMultimodal_1297080124022
// MI455X (gfx1250) — matrix operand order repaired and checked by reading, not compiled, not run on hardware
//
#include <hip/hip_runtime.h>
#include <hip/hip_bf16.h>
#include <stdint.h>

// ---------------------------------------------------------------------------
// AsyncNaiveMultimodal on gfx1250 (MI455X): 4 LSTMs + collapsed fusion head.
// Phase 0: x f32 -> bf16, zero-padded rows, (b,t)->(t,b) row transpose
// Phase 1: bf16 WMMA input GEMMs, C fragments stored packed (b128 I/O only)
// Phase 2: persistent per-modality scan, h/c in LDS, WMMA recurrent GEMM
// Phase 3: fusion collapsed to rank-1 dot with w_eff = W2*W1
// ---------------------------------------------------------------------------

typedef __attribute__((ext_vector_type(16))) __bf16 v16bf;
typedef __attribute__((ext_vector_type(8)))  float  v8f;

#define BB   64
#define TT   512
#define MTOT (BB * TT)

__device__ __forceinline__ unsigned short f2bf(float f) {
    unsigned int u = __float_as_uint(f);
    u = u + 0x7FFFu + ((u >> 16) & 1u);          // round-to-nearest-even
    return (unsigned short)(u >> 16);
}
__device__ __forceinline__ float bf2f(unsigned int bits16) {
    return __uint_as_float(bits16 << 16);
}
__device__ __forceinline__ unsigned int pack2bf(float a, float b) {
    return (unsigned int)f2bf(a) | ((unsigned int)f2bf(b) << 16);
}
__device__ __forceinline__ float sigmf(float x) {
    return 1.0f / (1.0f + __expf(-x));
}

union Frag {
    unsigned int u[8];
    v16bf        v;
    uint4        q[2];
};

// ---------------------------------------------------------------------------
// Phase 0: xbf[(t*64+b)*Dp + k] = bf16(x[b][t][k]), zero pad for k >= D.
// One thread per output dword (2 bf16).
// ---------------------------------------------------------------------------
__global__ void cvt_x_bf16(const float* __restrict__ x, int D, int Dp,
                           unsigned int* __restrict__ xbf, int totalDw) {
    int id = blockIdx.x * blockDim.x + threadIdx.x;
    if (id >= totalDw) return;
    int dwPerRow = Dp >> 1;
    int row = id / dwPerRow;
    int kk  = (id - row * dwPerRow) * 2;
    int b = row & 63, t = row >> 6;
    unsigned int v = 0u;
    if (kk < D) {
        float2 xv = *(const float2*)(x + (size_t)(b * TT + t) * D + kk);
        v = pack2bf(xv.x, xv.y);
    }
    xbf[id] = v;
}

// ---------------------------------------------------------------------------
// Prep: pack a [4H x K] f32 weight matrix (used as B = W^T, K x 4Hp, zero
// padded) into per-16x16x32-tile, lane-major bf16 dwords:
//   dst[tile*256 + lane*8 + r] = pack(B[k][n], B[k+1][n]),  n = nt*16+lane%16,
//   k = kt*32 + 8*(lane/16) + 2r + 8*(r/4)
// ---------------------------------------------------------------------------
__global__ void pack_b_bf16(const float* __restrict__ W, int Hq, int Hp,
                            int K, int Kp, unsigned int* __restrict__ dst,
                            int total) {
    int e = blockIdx.x * blockDim.x + threadIdx.x;
    if (e >= total) return;
    int r    = e & 7;
    int lane = (e >> 3) & 31;
    int tile = e >> 8;
    int KT = Kp >> 5;
    int nt = tile / KT;
    int kt = tile - nt * KT;
    int k  = kt * 32 + (lane >> 4) * 8 + 2 * r + (r >> 2) * 8;
    int n0 = nt * 16 + (lane & 15);
    float v0 = 0.f, v1 = 0.f;
    int g0 = n0 / Hp, j0 = n0 - g0 * Hp;
    if (j0 < Hq && k < K) v0 = W[(size_t)(g0 * Hq + j0) * K + k];
    if (j0 < Hq && k + 1 < K) v1 = W[(size_t)(g0 * Hq + j0) * K + k + 1];
    dst[e] = pack2bf(v0, v1);
}

// ---------------------------------------------------------------------------
// Prep: fusion head collapse.  w_eff[j] (padded-concat, 352) + b_eff at [352].
// ---------------------------------------------------------------------------
__global__ void build_weff(const float* __restrict__ W1, const float* __restrict__ b1,
                           const float* __restrict__ W2, const float* __restrict__ b2,
                           float* __restrict__ weff) {
    int j = threadIdx.x;
    if (j < 352) {
        int po, oo, H;
        if      (j < 128) { po = 0;   oo = 0;   H = 128; }
        else if (j < 160) { po = 128; oo = 128; H = 20;  }
        else if (j < 224) { po = 160; oo = 148; H = 64;  }
        else              { po = 224; oo = 212; H = 128; }
        int jp = j - po;
        float acc = 0.f;
        if (jp < H) {
            int col = oo + jp;
            for (int k = 0; k < 64; ++k) acc += W2[k] * W1[k * 340 + col];
        }
        weff[j] = acc;
    } else if (j == 352) {
        float acc = b2[0];
        for (int k = 0; k < 64; ++k) acc += W2[k] * b1[k];
        weff[352] = acc;
    }
}

// ---------------------------------------------------------------------------
// Phase 1: packed xg fragments.
//   xgP[(mtile*NT + ntile)*256 + lane*8 + r] = (bf16(x)*W_ih^T)[.] + bias(n)
// M rows ordered (t*64+b) so mtile = t*4 + mt matches the scan's tiles.
// A fragments: two b128 loads/lane from padded bf16 rows; B: two b128 loads.
// ---------------------------------------------------------------------------
__global__ void xg_gemm_bf16(const unsigned int* __restrict__ xbf, int Dp,
                             const unsigned int* __restrict__ wb, int KT, int NT,
                             const float* __restrict__ bih,
                             const float* __restrict__ bhh,
                             int H, int Hp, float* __restrict__ xgP) {
    int wave  = threadIdx.x >> 5;
    int lane  = threadIdx.x & 31;
    int mtile = blockIdx.x * 8 + wave;
    int ntile = blockIdx.y;
    int half  = lane >> 4;
    int lr    = lane & 15;

    const unsigned int* arow = xbf + (size_t)(mtile * 16 + lr) * (Dp >> 1);

    v8f acc;
#pragma unroll
    for (int r = 0; r < 8; ++r) acc[r] = 0.f;

    for (int kt = 0; kt < KT; ++kt) {
        Frag a, b;
        const uint4* ap = (const uint4*)(arow + kt * 16 + half * 4);
        a.q[0] = ap[0];
        a.q[1] = ap[2];                        // +16 bf16 elements
        const uint4* bp = (const uint4*)(wb + (((size_t)(ntile * KT + kt)) << 8) + lane * 8);
        b.q[0] = bp[0];
        b.q[1] = bp[1];
        acc = __builtin_amdgcn_wmma_f32_16x16x32_bf16(false, a.v, false, b.v,
                                                      (short)0, acc, false, false);
    }
    int n = ntile * 16 + lr;
    int g = n / Hp, jp = n - g * Hp;
    float bias = (jp < H) ? (bih[g * H + jp] + bhh[g * H + jp]) : 0.f;
    float4 o0 = make_float4(acc[0] + bias, acc[1] + bias, acc[2] + bias, acc[3] + bias);
    float4 o1 = make_float4(acc[4] + bias, acc[5] + bias, acc[6] + bias, acc[7] + bias);
    float* dst = xgP + (((size_t)mtile * NT + ntile) << 8) + lane * 8;
    *(float4*)(dst)     = o0;
    *(float4*)(dst + 4) = o1;
}

// ---------------------------------------------------------------------------
// Phase 2: persistent LSTM scan.  One block per modality, 32 waves.
// h (bf16) and c (f32) live in LDS; W_hh fragments stream from L2.
// Wave w owns (mt = w&3, jt = w>>2): all four gate tiles for rows
// mt*16..+15, cols jt*16..+15 -> local pointwise update, no cross-wave data.
// ---------------------------------------------------------------------------
struct ScanParams {
    const float*        xgP[4];
    const unsigned int* whh[4];
    const int*          pres[4];
    unsigned short*     hall[4];
    int H[4], Hp[4];
};

__global__ void lstm_scan_kernel(ScanParams P) {
    __shared__ float          ldsC[64 * 128];
    __shared__ unsigned short ldsH[64 * 128];

    int mod = blockIdx.x;
    const float*        xgP  = P.xgP[mod];
    const unsigned int* wb   = P.whh[mod];
    const int*          pres = P.pres[mod];
    unsigned short*     hall = P.hall[mod];
    int H  = P.H[mod];
    int Hp = P.Hp[mod];
    int JT = Hp >> 4;
    int KT = Hp >> 5;
    int NT = 4 * JT;

    int wave = threadIdx.x >> 5;
    int lane = threadIdx.x & 31;
    int half = lane >> 4;
    int lr   = lane & 15;

    for (int i = threadIdx.x; i < 64 * 128; i += blockDim.x) {
        ldsC[i] = 0.f;
        ldsH[i] = 0;
    }
    __syncthreads();

    int  pairs  = 4 * JT;
    bool active = wave < pairs;
    int  mt = wave & 3;
    int  jt = wave >> 2;

    for (int t = 0; t < TT; ++t) {
        __syncthreads();                       // step t-1 h visible everywhere
        Frag aF[4];
        if (active) {
            int m = mt * 16 + lr;
            for (int kt = 0; kt < KT; ++kt) {
                const uint4* hp = (const uint4*)&ldsH[m * 128 + kt * 32 + half * 8];
                aF[kt].q[0] = hp[0];
                aF[kt].q[1] = hp[2];           // +16 bf16 elements
            }
        }
        __syncthreads();                       // all A-frag reads done

        if (active) {
            v8f acc[4];
            for (int g = 0; g < 4; ++g) {
                const float4* cp = (const float4*)(xgP +
                    (((size_t)((t * 4 + mt) * NT + (g * JT + jt))) << 8) + lane * 8);
                float4 c0 = cp[0], c1 = cp[1];
                v8f c;
                c[0] = c0.x; c[1] = c0.y; c[2] = c0.z; c[3] = c0.w;
                c[4] = c1.x; c[5] = c1.y; c[6] = c1.z; c[7] = c1.w;
                for (int kt = 0; kt < KT; ++kt) {
                    Frag bfr;
                    const uint4* bp = (const uint4*)(wb +
                        (((size_t)((g * JT + jt) * KT + kt)) << 8) + lane * 8);
                    bfr.q[0] = bp[0];
                    bfr.q[1] = bp[1];
                    c = __builtin_amdgcn_wmma_f32_16x16x32_bf16(
                            false, aF[kt].v, false, bfr.v, (short)0, c, false, false);
                }
                acc[g] = c;
            }
#pragma unroll
            for (int r = 0; r < 8; ++r) {
                int b = mt * 16 + r + half * 8;
                int j = jt * 16 + lr;
                float iv = sigmf(acc[0][r]);
                float fv = sigmf(acc[1][r]);
                float gv = tanhf(acc[2][r]);
                float ov = sigmf(acc[3][r]);
                float cold = ldsC[b * 128 + j];
                float hold = bf2f(ldsH[b * 128 + j]);
                float cn = fv * cold + iv * gv;
                float hn = ov * tanhf(cn);
                float p  = (float)pres[b * TT + t];
                float c2 = p * cn + (1.f - p) * cold;
                float h2 = p * hn + (1.f - p) * hold;
                if (j < H) {
                    ldsC[b * 128 + j] = c2;
                    unsigned short hb = f2bf(h2);
                    ldsH[b * 128 + j] = hb;
                    hall[((size_t)(b * TT + t)) * Hp + j] = hb;
                }
            }
        }
    }
}

// ---------------------------------------------------------------------------
// Phase 3: out[b,t] = (t < seq[b]) ? sum_j h[b,t,j]*w_eff[j] + b_eff : 0
// ---------------------------------------------------------------------------
struct FuseParams {
    const unsigned short* hall[4];
    int H[4], Hp[4], po[4];
};

__global__ void fuse_kernel(FuseParams P, const float* __restrict__ weff,
                            const int* __restrict__ seq, float* __restrict__ out) {
    int idx = blockIdx.x * blockDim.x + threadIdx.x;
    if (idx >= MTOT) return;
    int b = idx >> 9;
    int t = idx & 511;
    if (t >= seq[b]) { out[idx] = 0.f; return; }
    float acc = weff[352];                     // b_eff
    for (int m = 0; m < 4; ++m) {
        const unsigned int* hm = (const unsigned int*)(P.hall[m] + (size_t)idx * P.Hp[m]);
        const float* wv = weff + P.po[m];
        int Hd = P.H[m] >> 1;                  // H even for all modalities
        for (int jd = 0; jd < Hd; ++jd) {
            unsigned int hv = hm[jd];
            acc += bf2f(hv & 0xFFFFu) * wv[2 * jd] + bf2f(hv >> 16) * wv[2 * jd + 1];
        }
    }
    out[idx] = acc;
}

// ---------------------------------------------------------------------------
extern "C" void kernel_launch(void* const* d_in, const int* in_sizes, int n_in,
                              void* d_out, int out_size, void* d_ws, size_t ws_size,
                              hipStream_t stream) {
    (void)in_sizes; (void)n_in; (void)out_size; (void)ws_size;
    static const int Hs[4]  = {128, 20, 64, 128};
    static const int Hps[4] = {128, 32, 64, 128};
    static const int Ds[4]  = {300, 30, 88, 1000};
    static const int Dps[4] = {320, 32, 96, 1024};

    const float* xin[4];  const int* pres[4];
    const float* Wih[4];  const float* Whh[4];
    const float* bih[4];  const float* bhh[4];
    for (int m = 0; m < 4; ++m) {
        xin[m]  = (const float*)d_in[6 * m + 0];
        pres[m] = (const int*)  d_in[6 * m + 1];
        Wih[m]  = (const float*)d_in[6 * m + 2];
        Whh[m]  = (const float*)d_in[6 * m + 3];
        bih[m]  = (const float*)d_in[6 * m + 4];
        bhh[m]  = (const float*)d_in[6 * m + 5];
    }
    const int*   seq = (const int*)  d_in[24];
    const float* fW1 = (const float*)d_in[26];
    const float* fb1 = (const float*)d_in[27];
    const float* fW2 = (const float*)d_in[28];
    const float* fb2 = (const float*)d_in[29];

    // workspace carve-up
    char* base = (char*)d_ws;
    size_t off = 0;
    unsigned int* pIH[4]; unsigned int* pHH[4]; unsigned int* xbf[4];
    float* xgP[4]; unsigned short* hall[4];
    int dwIH[4], dwHH[4], dwX[4];
    for (int m = 0; m < 4; ++m) {
        int Gp = 4 * Hps[m];
        dwIH[m] = (Gp / 16) * (Dps[m] / 32) * 256;
        dwHH[m] = (Gp / 16) * (Hps[m] / 32) * 256;
        dwX[m]  = MTOT * (Dps[m] / 2);
    }
    for (int m = 0; m < 4; ++m) { pIH[m] = (unsigned int*)(base + off); off += (size_t)dwIH[m] * 4; }
    for (int m = 0; m < 4; ++m) { pHH[m] = (unsigned int*)(base + off); off += (size_t)dwHH[m] * 4; }
    for (int m = 0; m < 4; ++m) { xbf[m] = (unsigned int*)(base + off); off += (size_t)dwX[m] * 4; }
    for (int m = 0; m < 4; ++m) { xgP[m] = (float*)(base + off);        off += (size_t)MTOT * 4 * Hps[m] * 4; }
    float* weff = (float*)(base + off); off += 356 * 4;
    for (int m = 0; m < 4; ++m) { hall[m] = (unsigned short*)(base + off); off += (size_t)MTOT * Hps[m] * 2; }

    // prep: x conversion, weight packing, fusion collapse
    for (int m = 0; m < 4; ++m) {
        cvt_x_bf16<<<(dwX[m] + 255) / 256, 256, 0, stream>>>(
            xin[m], Ds[m], Dps[m], xbf[m], dwX[m]);
        pack_b_bf16<<<(dwIH[m] + 255) / 256, 256, 0, stream>>>(
            Wih[m], Hs[m], Hps[m], Ds[m], Dps[m], pIH[m], dwIH[m]);
        pack_b_bf16<<<(dwHH[m] + 255) / 256, 256, 0, stream>>>(
            Whh[m], Hs[m], Hps[m], Hs[m], Hps[m], pHH[m], dwHH[m]);
    }
    build_weff<<<1, 512, 0, stream>>>(fW1, fb1, fW2, fb2, weff);

    // phase 1: input GEMMs (fragment-packed output)
    for (int m = 0; m < 4; ++m) {
        int NT = 4 * Hps[m] / 16;
        dim3 g(MTOT / 16 / 8, NT);
        xg_gemm_bf16<<<g, 256, 0, stream>>>(xbf[m], Dps[m], pIH[m], Dps[m] / 32, NT,
                                            bih[m], bhh[m], Hs[m], Hps[m], xgP[m]);
    }

    // phase 2: scans (one persistent block per modality)
    ScanParams sp;
    for (int m = 0; m < 4; ++m) {
        sp.xgP[m] = xgP[m]; sp.whh[m] = pHH[m]; sp.pres[m] = pres[m];
        sp.hall[m] = hall[m]; sp.H[m] = Hs[m]; sp.Hp[m] = Hps[m];
    }
    lstm_scan_kernel<<<4, 1024, 0, stream>>>(sp);

    // phase 3: collapsed fusion head
    FuseParams fp;
    const int po[4] = {0, 128, 160, 224};
    for (int m = 0; m < 4; ++m) {
        fp.hall[m] = hall[m]; fp.H[m] = Hs[m]; fp.Hp[m] = Hps[m]; fp.po[m] = po[m];
    }
    fuse_kernel<<<(MTOT + 255) / 256, 256, 0, stream>>>(fp, weff, seq, (float*)d_out);
}
